// MultiHeadAttention_63642825392306
// MI455X (gfx1250) — compile-verified, matrix operand order checked by reading, not run on hardware
//
#include <hip/hip_runtime.h>
#include <hip/hip_bf16.h>
#include <stdint.h>

// Attention forward for MI455X (gfx1250, wave32, WMMA).
// D = 1024, N = 4096. All GEMMs run through v_wmma_f32_16x16x32_bf16
// (f32 accumulate); softmax and epilogues in f32.  bf16 tile staging uses
// GLOBAL_LOAD_ASYNC_TO_LDS_B128 when the toolchain exposes the builtin.

#define DMODEL 1024
#define SEQ    4096

typedef __attribute__((ext_vector_type(16))) __bf16 v16bf;
typedef __attribute__((ext_vector_type(8)))  float  v8f;
typedef __attribute__((ext_vector_type(4)))  int    v4i;

#define BM 128
#define BN 128
#define BK 64
#define LDS_STRIDE 72   // ushorts per LDS row: 64 data + 8 pad (144B, 16B aligned)

#if defined(__has_builtin)
#if __has_builtin(__builtin_amdgcn_global_load_async_to_lds_b128)
#define HAVE_ASYNC_LDS 1
#endif
#endif
#ifndef HAVE_ASYNC_LDS
#define HAVE_ASYNC_LDS 0
#endif

#define AS_GLOBAL __attribute__((address_space(1)))
#define AS_LDS    __attribute__((address_space(3)))

__device__ __forceinline__ void wait_async_lds() {
#if HAVE_ASYNC_LDS
#if __has_builtin(__builtin_amdgcn_s_wait_asynccnt)
  __builtin_amdgcn_s_wait_asynccnt(0);
#else
  asm volatile("s_wait_asynccnt 0" ::: "memory");
#endif
#endif
}

struct Frag32 {
  unsigned int w0, w1, w2, w3, w4, w5, w6, w7;
};

// Native bf16 convert (lowered to v_cvt_*bf16_f32, packed when paired).
__device__ __forceinline__ unsigned short f2bf(float f) {
  __bf16 h = (__bf16)f;
  return __builtin_bit_cast(unsigned short, h);
}

__device__ __forceinline__ unsigned int pack2bf(float a, float b) {
  return (unsigned int)f2bf(a) | ((unsigned int)f2bf(b) << 16);
}

// Load one 16x32 bf16 operand fragment from LDS per CDNA5 16-bit A/B layout:
// lanes 0-15 hold row (lane), K = 0..7 and 16..23; lanes 16-31 hold row
// (lane-16), K = 8..15 and 24..31.  Two 16-byte ds_load_b128 per fragment.
__device__ __forceinline__ v16bf load_frag(const unsigned short* lds, int rowBase,
                                           int kstep, int lane) {
  const int row = rowBase + (lane & 15);
  const int k0  = kstep * 32 + (lane >> 4) * 8;
  const unsigned short* p = lds + row * LDS_STRIDE + k0;
  const uint4 lo = *(const uint4*)(p);
  const uint4 hi = *(const uint4*)(p + 16);
  Frag32 t{lo.x, lo.y, lo.z, lo.w, hi.x, hi.y, hi.z, hi.w};
  return __builtin_bit_cast(v16bf, t);
}

// Stage a 128x64 tile from a K-contiguous (row-major over K) source into LDS
// as bf16.  SRC_F32 converts f32 -> bf16 on the fly (packed converts);
// bf16 sources go through the async global->LDS path when available.
template<bool SRC_F32>
__device__ __forceinline__ void stage_rowmajor(unsigned short* lds, const void* src,
                                               int row0, int k0, int ld, int tid) {
  if (SRC_F32) {
    const float* s = (const float*)src;
#pragma unroll
    for (int it = 0; it < 8; ++it) {
      const int i = tid * 4 + it * 1024;          // 8192 f32 total
      const int r = i >> 6;
      const int c = i & 63;
      const float4 f = *(const float4*)(s + (size_t)(row0 + r) * ld + k0 + c);
      unsigned int* d = (unsigned int*)(lds + r * LDS_STRIDE + c);
      d[0] = pack2bf(f.x, f.y);
      d[1] = pack2bf(f.z, f.w);
    }
  } else {
    const unsigned short* s = (const unsigned short*)(src);
#pragma unroll
    for (int it = 0; it < 4; ++it) {
      const int i = tid * 8 + it * 2048;          // 8192 bf16 total
      const int r = i >> 6;
      const int c = i & 63;
      const unsigned short* gp = s + (size_t)(row0 + r) * ld + k0 + c;
      unsigned short* lp = lds + r * LDS_STRIDE + c;
#if HAVE_ASYNC_LDS
      __builtin_amdgcn_global_load_async_to_lds_b128(
          (AS_GLOBAL v4i*)(const void*)gp, (AS_LDS v4i*)(void*)lp, 0, 0);
#else
      *(uint4*)lp = *(const uint4*)gp;
#endif
    }
  }
}

// Stage a 64(K) x 128(N) bf16 tile from an N-contiguous source (vp), storing
// transposed into LDS so downstream fragment loads see the N-major layout.
__device__ __forceinline__ void stage_b_trans(unsigned short* lds, const unsigned short* src,
                                              int k0, int n0, int ldb, int tid) {
#pragma unroll
  for (int it = 0; it < 4; ++it) {
    const int i  = tid * 8 + it * 2048;           // 8192 bf16 total
    const int kk = i >> 7;
    const int nn = i & 127;
    const uint4 u = *(const uint4*)(src + (size_t)(k0 + kk) * ldb + n0 + nn);
    const unsigned int w[4] = {u.x, u.y, u.z, u.w};
#pragma unroll
    for (int j = 0; j < 4; ++j) {
      lds[(nn + 2 * j + 0) * LDS_STRIDE + kk] = (unsigned short)(w[j] & 0xFFFFu);
      lds[(nn + 2 * j + 1) * LDS_STRIDE + kk] = (unsigned short)(w[j] >> 16);
    }
  }
}

// C[M,Ncols] = A[M,K] * B^T (+ bias).  B_TRANS=true: B stored [Ncols,K]
// (K-contiguous, natural A*B^T).  B_TRANS=false: B stored [K,Ncols] (bf16
// only, transposed during LDS staging).  Block tile 128x128x64, 8 waves,
// each wave owns a 32x64 patch = 2x4 WMMA fragments, 16 WMMAs per K-stage.
template<bool A_F32, bool B_F32, bool B_TRANS, bool OUT_F32, bool BIAS>
__global__ __launch_bounds__(256)
void gemm_bf16_wmma(const void* __restrict__ A, const void* __restrict__ B,
                    const float* __restrict__ bias, void* __restrict__ C,
                    int K, int lda, int ldb, int ldc) {
  __shared__ __align__(16) unsigned short ldsA[BM * LDS_STRIDE];
  __shared__ __align__(16) unsigned short ldsB[BN * LDS_STRIDE];

  const int tid  = threadIdx.x;
  const int lane = tid & 31;
  const int wave = tid >> 5;
  const int wm   = wave & 3;     // 4 waves tiling M (32 rows each)
  const int wn   = wave >> 2;    // 2 waves tiling N (64 cols each)
  const int blockN0 = blockIdx.x * BN;
  const int blockM0 = blockIdx.y * BM;

  const v8f vzero = {0.f, 0.f, 0.f, 0.f, 0.f, 0.f, 0.f, 0.f};
  v8f acc[2][4];
#pragma unroll
  for (int i = 0; i < 2; ++i)
#pragma unroll
    for (int j = 0; j < 4; ++j) acc[i][j] = vzero;

  for (int k0 = 0; k0 < K; k0 += BK) {
    // Warm L2->WGP path for the next K-tile while we work on this one.
    if (k0 + BK < K) {
      const size_t aelem = (size_t)(blockM0 + (tid >> 1)) * lda + (k0 + BK) + (tid & 1) * 32;
      if (A_F32) __builtin_prefetch((const float*)A + aelem, 0, 3);
      else       __builtin_prefetch((const unsigned short*)A + aelem, 0, 3);
      if (B_TRANS) {
        const size_t belem = (size_t)(blockN0 + (tid >> 1)) * ldb + (k0 + BK) + (tid & 1) * 32;
        if (B_F32) __builtin_prefetch((const float*)B + belem, 0, 3);
        else       __builtin_prefetch((const unsigned short*)B + belem, 0, 3);
      } else {
        const size_t belem = (size_t)(k0 + BK + (tid >> 2)) * ldb + blockN0 + (tid & 3) * 32;
        __builtin_prefetch((const unsigned short*)B + belem, 0, 3);
      }
    }

    stage_rowmajor<A_F32>(ldsA, A, blockM0, k0, lda, tid);
    if (B_TRANS) {
      stage_rowmajor<B_F32>(ldsB, B, blockN0, k0, ldb, tid);
    } else {
      stage_b_trans(ldsB, (const unsigned short*)B, k0, blockN0, ldb, tid);
    }
    if (!A_F32 || (B_TRANS && !B_F32)) wait_async_lds();
    __syncthreads();

#pragma unroll
    for (int ks = 0; ks < 2; ++ks) {
      v16bf afrag[2], bfrag[4];
#pragma unroll
      for (int mi = 0; mi < 2; ++mi) afrag[mi] = load_frag(ldsA, wm * 32 + mi * 16, ks, lane);
#pragma unroll
      for (int ni = 0; ni < 4; ++ni) bfrag[ni] = load_frag(ldsB, wn * 64 + ni * 16, ks, lane);

#pragma unroll
      for (int mi = 0; mi < 2; ++mi)
#pragma unroll
        for (int ni = 0; ni < 4; ++ni)
          acc[mi][ni] = __builtin_amdgcn_wmma_f32_16x16x32_bf16(
              false, afrag[mi], false, bfrag[ni], (short)0, acc[mi][ni], false, false);
    }
    __syncthreads();
  }

  // Epilogue: C/D layout -> VGPR r holds M = r (+8 for lanes 16-31), N = lane&15.
  const int nlane = lane & 15;
  const int moff  = (lane >> 4) * 8;
#pragma unroll
  for (int mi = 0; mi < 2; ++mi) {
#pragma unroll
    for (int ni = 0; ni < 4; ++ni) {
      const int col = blockN0 + wn * 64 + ni * 16 + nlane;
      const float b = BIAS ? bias[col] : 0.f;
#pragma unroll
      for (int r = 0; r < 8; ++r) {
        const int row = blockM0 + wm * 32 + mi * 16 + moff + r;
        const float val = acc[mi][ni][r] + b;
        if (OUT_F32) ((float*)C)[(size_t)row * ldc + col] = val;
        else         ((unsigned short*)C)[(size_t)row * ldc + col] = f2bf(val);
      }
    }
  }
}

// One block per row; 256 threads * 16 contiguous f32 = 4096 cols.
__global__ __launch_bounds__(256)
void softmax_rows(float* __restrict__ attn, int n) {
  const int row  = blockIdx.x;
  const int tid  = threadIdx.x;
  const int lane = tid & 31;
  const int wave = tid >> 5;
  float* p = attn + (size_t)row * n;
  __shared__ float red[8];

  float v[16];
#pragma unroll
  for (int it = 0; it < 4; ++it) {
    const float4 f = *(const float4*)(p + tid * 16 + it * 4);
    v[it * 4 + 0] = f.x; v[it * 4 + 1] = f.y;
    v[it * 4 + 2] = f.z; v[it * 4 + 3] = f.w;
  }

  float m = v[0];
#pragma unroll
  for (int i = 1; i < 16; ++i) m = fmaxf(m, v[i]);
#pragma unroll
  for (int off = 16; off > 0; off >>= 1) m = fmaxf(m, __shfl_xor(m, off, 32));
  if (lane == 0) red[wave] = m;
  __syncthreads();
  float rmax = red[0];
#pragma unroll
  for (int i = 1; i < 8; ++i) rmax = fmaxf(rmax, red[i]);
  __syncthreads();

  float s = 0.f;
#pragma unroll
  for (int i = 0; i < 16; ++i) { v[i] = __expf(v[i] - rmax); s += v[i]; }
#pragma unroll
  for (int off = 16; off > 0; off >>= 1) s += __shfl_xor(s, off, 32);
  if (lane == 0) red[wave] = s;
  __syncthreads();
  float rsum = 0.f;
#pragma unroll
  for (int i = 0; i < 8; ++i) rsum += red[i];
  const float inv = 1.0f / rsum;

#pragma unroll
  for (int it = 0; it < 4; ++it) {
    float4 f;
    f.x = v[it * 4 + 0] * inv; f.y = v[it * 4 + 1] * inv;
    f.z = v[it * 4 + 2] * inv; f.w = v[it * 4 + 3] * inv;
    *(float4*)(p + tid * 16 + it * 4) = f;
  }
}

extern "C" void kernel_launch(void* const* d_in, const int* in_sizes, int n_in,
                              void* d_out, int out_size, void* d_ws, size_t ws_size,
                              hipStream_t stream) {
  (void)in_sizes; (void)n_in; (void)out_size; (void)ws_size;

  const float* q    = (const float*)d_in[0];
  const float* k    = (const float*)d_in[1];
  const float* v    = (const float*)d_in[2];
  const float* wq_w = (const float*)d_in[3];
  const float* wq_b = (const float*)d_in[4];
  const float* wk_w = (const float*)d_in[5];
  const float* wk_b = (const float*)d_in[6];
  const float* wv_w = (const float*)d_in[7];
  const float* wv_b = (const float*)d_in[8];
  const float* wo_w = (const float*)d_in[9];
  const float* wo_b = (const float*)d_in[10];

  float* out_x    = (float*)d_out;                       // [N, D]
  float* out_attn = out_x + (size_t)SEQ * DMODEL;        // [N, N]

  unsigned short* ws = (unsigned short*)d_ws;            // bf16 scratch
  unsigned short* qp = ws;                               // [N, D] bf16
  unsigned short* kp = ws + 1 * (size_t)SEQ * DMODEL;    // [N, D] bf16
  unsigned short* vp = ws + 2 * (size_t)SEQ * DMODEL;    // [N, D] bf16
  unsigned short* x1 = ws + 3 * (size_t)SEQ * DMODEL;    // [N, D] bf16

  const dim3 blk(256);
  const dim3 gProj(DMODEL / BN, SEQ / BM);   // (8, 32)
  const dim3 gScores(SEQ / BN, SEQ / BM);    // (32, 32)

  // qp/kp/vp = X @ W^T + b  (f32 in, bf16 out)
  gemm_bf16_wmma<true, true, true, false, true><<<gProj, blk, 0, stream>>>(
      q, wq_w, wq_b, qp, DMODEL, DMODEL, DMODEL, DMODEL);
  gemm_bf16_wmma<true, true, true, false, true><<<gProj, blk, 0, stream>>>(
      k, wk_w, wk_b, kp, DMODEL, DMODEL, DMODEL, DMODEL);
  gemm_bf16_wmma<true, true, true, false, true><<<gProj, blk, 0, stream>>>(
      v, wv_w, wv_b, vp, DMODEL, DMODEL, DMODEL, DMODEL);

  // scores = qp @ kp^T  (bf16 x bf16 -> f32 into d_out attention region)
  gemm_bf16_wmma<false, false, true, true, false><<<gScores, blk, 0, stream>>>(
      qp, kp, nullptr, out_attn, DMODEL, DMODEL, DMODEL, SEQ);

  // softmax rows in place
  softmax_rows<<<SEQ, blk, 0, stream>>>(out_attn, SEQ);

  // x1 = attn @ vp  (f32 attn converted on stage; vp transposed during staging)
  gemm_bf16_wmma<true, false, false, false, false><<<gProj, blk, 0, stream>>>(
      out_attn, vp, nullptr, x1, SEQ, SEQ, DMODEL, DMODEL);

  // x = x1 @ wo^T + b  (bf16 x f32-converted -> f32 out)
  gemm_bf16_wmma<false, true, true, true, true><<<gProj, blk, 0, stream>>>(
      x1, wo_w, wo_b, out_x, DMODEL, DMODEL, DMODEL, DMODEL);
}
